// BasicNCA3D_17068200034549
// MI455X (gfx1250) — matrix operand order repaired and checked by reading, not compiled, not run on hardware
//
#include <hip/hip_runtime.h>
#include <hip/hip_bf16.h>

typedef __attribute__((ext_vector_type(16))) _Float16 v16h;
typedef __attribute__((ext_vector_type(8)))  float    v8f;
typedef __attribute__((ext_vector_type(4)))  int      v4i;
typedef __attribute__((address_space(1))) v4i* gptr_v4i;
typedef __attribute__((address_space(3))) v4i* lptr_v4i;

#define DSP   48
#define NC    16
#define NH    128
#define NBATCH 2
#define NVOX  (NBATCH*DSP*DSP*DSP)   /* 221184 */
#define BLK   256
#define NBLK  (NVOX/BLK)             /* 864 */
#define KTAP  7
#define NTAPS (KTAP*KTAP*KTAP)       /* 343 */
#define PADH  (NH + 8)               /* 136 halves: break 256B-stride bank aliasing */

__device__ __forceinline__ int refl(int i) {
    return i < 0 ? -i : (i >= DSP ? 2 * DSP - 2 - i : i);
}

// global -> LDS 16B chunk copy; async-DMA path on gfx1250 when available.
__device__ __forceinline__ void g2l_b128(const _Float16* gsrc, _Float16* ldst) {
#if __has_builtin(__builtin_amdgcn_global_load_async_to_lds_b128)
    __builtin_amdgcn_global_load_async_to_lds_b128(
        (gptr_v4i)(v4i*)(uintptr_t)gsrc,
        (lptr_v4i)(v4i*)ldst, 0, 0);
#else
    *reinterpret_cast<uint4*>(ldst) = *reinterpret_cast<const uint4*>(gsrc);
#endif
}

__device__ __forceinline__ void g2l_wait() {
#if __has_builtin(__builtin_amdgcn_global_load_async_to_lds_b128)
#if __has_builtin(__builtin_amdgcn_s_wait_asynccnt)
    __builtin_amdgcn_s_wait_asynccnt(0);
#else
    asm volatile("s_wait_asynccnt 0x0" ::: "memory");
#endif
#endif
}

__global__ void k_zero_stats(float* g) { g[threadIdx.x] = 0.f; }

__global__ void k_bn_finalize(const float* __restrict__ stats,
                              const float* __restrict__ gamma,
                              const float* __restrict__ beta,
                              float* __restrict__ scsh) {
    int c = threadIdx.x;                 // 128 threads
    float N = (float)NVOX;
    float mean = stats[c] / N;
    float var  = stats[NH + c] / N - mean * mean;
    float r    = rsqrtf(var + 1e-5f);
    float sc   = gamma[c] * r;
    scsh[c]      = sc;
    scsh[NH + c] = beta[c] - mean * sc;
}

// Pass 1: depthwise 7^3 conv (reflect) + concat + fc0 GEMM via WMMA.
// h (f16) staged in LDS and written coalesced (b128); per-channel sum/sumsq.
__global__ void __launch_bounds__(BLK)
k_perceive_fc0(const float* __restrict__ xin,
               const float* __restrict__ conv_w,
               const float* __restrict__ conv_b,
               const float* __restrict__ fc0_w,
               const float* __restrict__ fc0_b,
               _Float16* __restrict__ hbuf,
               float* __restrict__ gstats) {
    __shared__ __align__(16) union {
        struct {
            float    w[NTAPS * NC];      // 21952 B
            _Float16 y[BLK][34];         // 17408 B  (A staging, padded)
        } p1;
        _Float16 out[BLK][NH];           // 65536 B  (h output tile)
    } smem;
    __shared__ float s_st[2 * NH];

    int tid = threadIdx.x;
    for (int i = tid; i < NTAPS * NC; i += BLK) smem.p1.w[i] = conv_w[i];
    if (tid < 2 * NH) s_st[tid] = 0.f;
    __syncthreads();

    int v  = blockIdx.x * BLK + tid;
    int b  = v / (DSP * DSP * DSP);
    int rr = v - b * (DSP * DSP * DSP);
    int ix = rr / (DSP * DSP);
    int iy = (rr / DSP) % DSP;
    int iz = rr % DSP;

    const float4* x4 = reinterpret_cast<const float4*>(xin);
    const float4* w4 = reinterpret_cast<const float4*>(smem.p1.w);

    float acc[NC];
#pragma unroll
    for (int c = 0; c < NC; ++c) acc[c] = 0.f;

    for (int kd = 0; kd < KTAP; ++kd) {
        int jx = refl(ix + kd - 3);
        for (int kh = 0; kh < KTAP; ++kh) {
            int jy = refl(iy + kh - 3);
#pragma unroll
            for (int kw = 0; kw < KTAP; ++kw) {
                int jz  = refl(iz + kw - 3);
                int jv  = ((b * DSP + jx) * DSP + jy) * DSP + jz;
                int tap = (kd * KTAP + kh) * KTAP + kw;
                const float4* xp = x4 + (size_t)jv * 4;
                const float4* wp = w4 + tap * 4;
#pragma unroll
                for (int q = 0; q < 4; ++q) {
                    float4 xv = xp[q];
                    float4 wv = wp[q];
                    acc[4 * q + 0] += xv.x * wv.x;
                    acc[4 * q + 1] += xv.y * wv.y;
                    acc[4 * q + 2] += xv.z * wv.z;
                    acc[4 * q + 3] += xv.w * wv.w;
                }
            }
        }
    }

    {   // y = [x, conv + conv_b] as f16 row in LDS
        const float4* xp = x4 + (size_t)v * 4;
#pragma unroll
        for (int q = 0; q < 4; ++q) {
            float4 xv = xp[q];
            smem.p1.y[tid][4 * q + 0] = (_Float16)xv.x;
            smem.p1.y[tid][4 * q + 1] = (_Float16)xv.y;
            smem.p1.y[tid][4 * q + 2] = (_Float16)xv.z;
            smem.p1.y[tid][4 * q + 3] = (_Float16)xv.w;
        }
#pragma unroll
        for (int c = 0; c < NC; ++c)
            smem.p1.y[tid][NC + c] = (_Float16)(acc[c] + conv_b[c]);
    }
    __syncthreads();

    int wave = tid >> 5, lane = tid & 31;
    int lrow = lane & 15, mhalf = lane >> 4;

    // B fragments: fc0_w [32,128] f32 -> f16, 8 N-tiles
    v16h bf[8];
#pragma unroll
    for (int nt = 0; nt < 8; ++nt) {
        int n = nt * 16 + lrow;
#pragma unroll
        for (int j = 0; j < 8; ++j) {
            int K = mhalf * 16 + 2 * j;
            bf[nt][2 * j]     = (_Float16)fc0_w[K * NH + n];
            bf[nt][2 * j + 1] = (_Float16)fc0_w[(K + 1) * NH + n];
        }
    }

    // A fragments for both M-tiles (pulled before union area is reused)
    v16h a[2];
#pragma unroll
    for (int mt = 0; mt < 2; ++mt) {
#pragma unroll
        for (int j = 0; j < 8; ++j) {
            int K = mhalf * 16 + 2 * j;
            a[mt][2 * j]     = smem.p1.y[wave * 32 + mt * 16 + lrow][K];
            a[mt][2 * j + 1] = smem.p1.y[wave * 32 + mt * 16 + lrow][K + 1];
        }
    }
    __syncthreads();   // everyone done reading p1.y; union now used as out-tile

#pragma unroll
    for (int mt = 0; mt < 2; ++mt) {
        int mbase = wave * 32 + mt * 16;
#pragma unroll
        for (int nt = 0; nt < 8; ++nt) {
            v8f cc = {};
            cc = __builtin_amdgcn_wmma_f32_16x16x32_f16(
                false, a[mt], false, bf[nt], (short)0, cc, false, false);
            int n = nt * 16 + lrow;
            float bias = fc0_b[n];
            float ls = 0.f, lq = 0.f;
#pragma unroll
            for (int j = 0; j < 8; ++j) {
                float hv = cc[j] + bias;
                smem.out[mbase + mhalf * 8 + j][n] = (_Float16)hv;
                ls += hv;
                lq += hv * hv;
            }
            atomicAdd(&s_st[n], ls);
            atomicAdd(&s_st[NH + n], lq);
        }
    }
    __syncthreads();

    // coalesced b128 store of the 256x128 f16 tile
    {
        const uint4* so = reinterpret_cast<const uint4*>(&smem.out[0][0]);
        uint4* go = reinterpret_cast<uint4*>(hbuf) + (size_t)blockIdx.x * (BLK * NH / 8);
#pragma unroll
        for (int i = 0; i < BLK * NH / 8 / BLK; ++i) {   // 16 iters
            int c = tid + i * BLK;
            go[c] = so[c];
        }
    }
    if (tid < 2 * NH) atomicAdd(&gstats[tid], s_st[tid]);
}

// Pass 2: h tile -> LDS (async b128), BN+ReLU at fragment build, fc1 WMMA,
// mask + residual (channel 0 frozen).
__global__ void __launch_bounds__(BLK)
k_bn_fc1_residual(const _Float16* __restrict__ hbuf,
                  const float* __restrict__ scsh,
                  const float* __restrict__ fc1_w,
                  const int* __restrict__ mask,
                  const float* __restrict__ xin,
                  float* __restrict__ xout) {
    __shared__ __align__(16) _Float16 s_h[BLK * PADH];   // 69632 B raw h tile
    __shared__ float s_sc[2 * NH];                        // BN scale/shift

    int tid = threadIdx.x, wave = tid >> 5, lane = tid & 31;
    int lrow = lane & 15, mhalf = lane >> 4;

    if (tid < 2 * NH) s_sc[tid] = scsh[tid];

    // stage 256x128 f16 h tile: global (coalesced b128) -> LDS (padded rows)
#pragma unroll
    for (int i = 0; i < BLK * NH / 8 / BLK; ++i) {        // 16 iters
        int c    = tid + i * BLK;                         // chunk id
        int vox  = c >> 4;                                // local voxel
        int part = c & 15;                                // 8-half chunk in row
        const _Float16* gsrc = hbuf + ((size_t)blockIdx.x * BLK * NH + (size_t)c * 8);
        g2l_b128(gsrc, &s_h[vox * PADH + part * 8]);
    }
    g2l_wait();
    __syncthreads();

    // B fragments: fc1_w [128,16] f32 -> f16, 4 K-chunks, single N-tile
    v16h bf[4];
#pragma unroll
    for (int kc = 0; kc < 4; ++kc) {
#pragma unroll
        for (int j = 0; j < 8; ++j) {
            int K = kc * 32 + mhalf * 16 + 2 * j;
            bf[kc][2 * j]     = (_Float16)fc1_w[K * NC + lrow];
            bf[kc][2 * j + 1] = (_Float16)fc1_w[(K + 1) * NC + lrow];
        }
    }

#pragma unroll
    for (int mt = 0; mt < 2; ++mt) {
        int lrowbase = wave * 32 + mt * 16 + lrow;        // local A row
        int arow = lrowbase * PADH;
        v8f cc = {};
#pragma unroll
        for (int kc = 0; kc < 4; ++kc) {
            v16h av;
#pragma unroll
            for (int j = 0; j < 8; ++j) {
                int K = kc * 32 + mhalf * 16 + 2 * j;
                float h0 = (float)s_h[arow + K];
                float h1 = (float)s_h[arow + K + 1];
                float v0 = fmaxf(h0 * s_sc[K]     + s_sc[NH + K],     0.f);
                float v1 = fmaxf(h1 * s_sc[K + 1] + s_sc[NH + K + 1], 0.f);
                av[2 * j]     = (_Float16)v0;
                av[2 * j + 1] = (_Float16)v1;
            }
            cc = __builtin_amdgcn_wmma_f32_16x16x32_f16(
                false, av, false, bf[kc], (short)0, cc, false, false);
        }
        int vg0 = blockIdx.x * BLK + wave * 32 + mt * 16 + mhalf * 8;
#pragma unroll
        for (int j = 0; j < 8; ++j) {
            int vg = vg0 + j;
            float m   = (float)mask[vg];
            float old = xin[(size_t)vg * NC + lrow];
            float dx  = cc[j] * m;
            xout[(size_t)vg * NC + lrow] = old + ((lrow == 0) ? 0.f : dx);
        }
    }
}

extern "C" void kernel_launch(void* const* d_in, const int* in_sizes, int n_in,
                              void* d_out, int out_size, void* d_ws, size_t ws_size,
                              hipStream_t stream) {
    const float* x      = (const float*)d_in[0];
    const float* conv_w = (const float*)d_in[1];
    const float* conv_b = (const float*)d_in[2];
    const float* fc0_w  = (const float*)d_in[3];
    const float* fc0_b  = (const float*)d_in[4];
    const float* fc1_w  = (const float*)d_in[5];
    const float* gamma  = (const float*)d_in[6];
    const float* beta   = (const float*)d_in[7];
    const int*   masks  = (const int*)d_in[8];
    float* out = (float*)d_out;
    (void)in_sizes; (void)n_in; (void)out_size; (void)ws_size;

    char* ws = (char*)d_ws;
    size_t off = 0;
    _Float16* hbuf = (_Float16*)(ws + off);
    off += (size_t)NVOX * NH * sizeof(_Float16);  off = (off + 255) & ~(size_t)255;
    float* X0 = (float*)(ws + off);
    off += (size_t)NVOX * NC * sizeof(float);     off = (off + 255) & ~(size_t)255;
    float* X1 = (float*)(ws + off);
    off += (size_t)NVOX * NC * sizeof(float);     off = (off + 255) & ~(size_t)255;
    float* gstats = (float*)(ws + off);
    off += 2 * NH * sizeof(float);
    float* scsh = (float*)(ws + off);

    float* ping[2] = {X0, X1};
    for (int s = 0; s < 8; ++s) {
        const float* xin_s = (s == 0) ? x : ping[(s - 1) & 1];
        float* xout_s      = (s == 7) ? out : ping[s & 1];
        k_zero_stats<<<1, 2 * NH, 0, stream>>>(gstats);
        k_perceive_fc0<<<NBLK, BLK, 0, stream>>>(xin_s, conv_w, conv_b,
                                                 fc0_w, fc0_b, hbuf, gstats);
        k_bn_finalize<<<1, NH, 0, stream>>>(gstats, gamma, beta, scsh);
        k_bn_fc1_residual<<<NBLK, BLK, 0, stream>>>(hbuf, scsh, fc1_w,
                                                    masks + (size_t)s * NVOX,
                                                    xin_s, xout_s);
    }
}
